// MultiHeadAttention_412316860641
// MI455X (gfx1250) — matrix operand order repaired and checked by reading, not compiled, not run on hardware
//
#include <hip/hip_runtime.h>
#include <hip/hip_bf16.h>

// ---------------- problem constants ----------------
constexpr int Bc = 4;
constexpr int Sc = 2048;
constexpr int Dc = 1024;
constexpr int Hc = 16;
constexpr int Ec = 64;
constexpr int BSc = Bc * Sc;        // 8192 rows
constexpr int BHc = Bc * Hc;        // 64 (b,h) pairs
constexpr float LN_EPS = 1e-5f;

// ---------------- vector types ----------------
typedef __attribute__((ext_vector_type(16))) _Float16 v16h;
typedef __attribute__((ext_vector_type(8)))  _Float16 v8h;
typedef __attribute__((ext_vector_type(4)))  _Float16 v4h;
typedef __attribute__((ext_vector_type(8)))  float    v8f;

union AF { v16h v; v8h h[2]; };

// D = A*B + C, A 16x32 f16, B 32x16 f16, C/D 16x16 f32
__device__ __forceinline__ v8f wmma_f16(v16h a, v16h b, v8f c) {
  return __builtin_amdgcn_wmma_f32_16x16x32_f16(
      /*neg_a=*/false, a, /*neg_b=*/false, b,
      /*c_mod=*/(short)0, c, /*reuse_a=*/false, /*reuse_b=*/false);
}

// ---------------- kernel 1: f32 -> f16 convert ----------------
__global__ void cvt_f32_to_f16(const float* __restrict__ in,
                               _Float16* __restrict__ out, int n) {
  int i = blockIdx.x * blockDim.x + threadIdx.x;
  if (i < n) out[i] = (_Float16)in[i];
}

// ---------------- kernel 2: layernorm (f32 stats) -> f16 ----------------
__global__ __launch_bounds__(256)
void ln_kernel(const float* __restrict__ x, const float* __restrict__ gamma,
               const float* __restrict__ beta, _Float16* __restrict__ xn) {
  int row = blockIdx.x;
  int tid = threadIdx.x;
  const float* xr = x + (size_t)row * Dc;
  float4 v = ((const float4*)xr)[tid];               // 256 * 4 = 1024
  float s  = v.x + v.y + v.z + v.w;
  float sq = v.x * v.x + v.y * v.y + v.z * v.z + v.w * v.w;

  __shared__ float ssum[256];
  __shared__ float ssq[256];
  ssum[tid] = s; ssq[tid] = sq;
  __syncthreads();
  for (int off = 128; off > 0; off >>= 1) {
    if (tid < off) { ssum[tid] += ssum[tid + off]; ssq[tid] += ssq[tid + off]; }
    __syncthreads();
  }
  float mean = ssum[0] * (1.0f / Dc);
  float var  = ssq[0] * (1.0f / Dc) - mean * mean;
  float rs   = rsqrtf(var + LN_EPS);

  float4 g = ((const float4*)gamma)[tid];
  float4 be = ((const float4*)beta)[tid];
  v4h o;
  o[0] = (_Float16)((v.x - mean) * rs * g.x + be.x);
  o[1] = (_Float16)((v.y - mean) * rs * g.y + be.y);
  o[2] = (_Float16)((v.z - mean) * rs * g.z + be.z);
  o[3] = (_Float16)((v.w - mean) * rs * g.w + be.w);
  *(v4h*)(xn + (size_t)row * Dc + tid * 4) = o;
}

// ---------------- kernel 3: QKV projection GEMM (WMMA) ----------------
// C[BS, D] = xn[BS, D] * W^T (W is [HE, D] row-major)  + bias
// each wave: 16x64 output tile. Store f16 to [BH,S,E] (transposed=0)
// or [BH,E,S] (transposed=1, used for V).
__global__ __launch_bounds__(256)
void proj_qkv(const _Float16* __restrict__ xn, const _Float16* __restrict__ W,
              const float* __restrict__ bias, _Float16* __restrict__ dst,
              int transposed) {
  int lane = threadIdx.x & 31;
  int wave = threadIdx.x >> 5;
  int tile = blockIdx.x * 8 + wave;        // 8192 tiles
  int m0 = (tile >> 4) * 16;               // 512 M tiles
  int n0 = (tile & 15) * 64;               // 16 N tiles
  int hv  = lane >> 4;                     // half-wave id
  int l16 = lane & 15;

  v8f acc[4] = {v8f{}, v8f{}, v8f{}, v8f{}};
  const _Float16* arow = xn + (size_t)(m0 + l16) * Dc;

  for (int k0 = 0; k0 < Dc; k0 += 32) {
    AF a;
    a.h[0] = *(const v8h*)(arow + k0 + hv * 8);
    a.h[1] = *(const v8h*)(arow + k0 + 16 + hv * 8);
#pragma unroll
    for (int ns = 0; ns < 4; ++ns) {
      AF bm;
      const _Float16* wr = W + (size_t)(n0 + ns * 16 + l16) * Dc + k0 + hv * 8;
      bm.h[0] = *(const v8h*)(wr);
      bm.h[1] = *(const v8h*)(wr + 16);
      acc[ns] = wmma_f16(a.v, bm.v, acc[ns]);
    }
  }
#pragma unroll
  for (int ns = 0; ns < 4; ++ns) {
    int ncol = n0 + ns * 16 + l16;
    float bsv = bias[ncol];
    int hh = ncol >> 6;        // head
    int e  = ncol & 63;
#pragma unroll
    for (int r = 0; r < 8; ++r) {
      int row = m0 + r + 8 * hv;
      int b = row / Sc, s = row % Sc;
      float val = acc[ns][r] + bsv;
      size_t idx = transposed
          ? ((size_t)((b * Hc + hh) * Ec + e) * Sc + s)
          : ((size_t)((b * Hc + hh) * Sc + s) * Ec + e);
      dst[idx] = (_Float16)val;
    }
  }
}

// ---------------- kernel 4: flash attention (WMMA + online softmax) -------
// one wave per (bh, 16-query block). q,k in [BH,S,E] f16, vT in [BH,E,S] f16.
__global__ __launch_bounds__(32)
void flash_attn(const _Float16* __restrict__ q, const _Float16* __restrict__ k,
                const _Float16* __restrict__ vT, _Float16* __restrict__ heads) {
  __shared__ alignas(16) _Float16 pbuf[16 * 32];   // P tile staging

  int lane = threadIdx.x;
  int hv  = lane >> 4;
  int l16 = lane & 15;
  int bh = blockIdx.x >> 7;              // S/16 = 128 query blocks
  int m0 = (blockIdx.x & 127) << 4;

  const _Float16* qb = q  + (size_t)bh * Sc * Ec;
  const _Float16* kb = k  + (size_t)bh * Sc * Ec;
  const _Float16* vb = vT + (size_t)bh * Ec * Sc;

  // Q A-fragments for K(=e) chunks [0,32) and [32,64)
  AF aq0, aq1;
  {
    const _Float16* qr = qb + (size_t)(m0 + l16) * Ec;
    aq0.h[0] = *(const v8h*)(qr + hv * 8);
    aq0.h[1] = *(const v8h*)(qr + 16 + hv * 8);
    aq1.h[0] = *(const v8h*)(qr + 32 + hv * 8);
    aq1.h[1] = *(const v8h*)(qr + 48 + hv * 8);
  }

  v8f O[4] = {v8f{}, v8f{}, v8f{}, v8f{}};
  float mrow[8], lrow[8];
#pragma unroll
  for (int r = 0; r < 8; ++r) { mrow[r] = -1e30f; lrow[r] = 0.0f; }
  const float sc = 0.125f;               // 1/(sqrt(E)*TEMP)

  for (int t0 = 0; t0 < Sc; t0 += 32) {
    // ---- scores S = Q * K^T for 32 keys (two 16-key N subtiles) ----
    v8f s0 = v8f{}, s1 = v8f{};
    {
      const _Float16* kr0 = kb + (size_t)(t0 + l16) * Ec + hv * 8;
      const _Float16* kr1 = kb + (size_t)(t0 + 16 + l16) * Ec + hv * 8;
      AF bk0, bk1;
      bk0.h[0] = *(const v8h*)(kr0);      bk0.h[1] = *(const v8h*)(kr0 + 16);
      bk1.h[0] = *(const v8h*)(kr1);      bk1.h[1] = *(const v8h*)(kr1 + 16);
      s0 = wmma_f16(aq0.v, bk0.v, s0);
      s1 = wmma_f16(aq0.v, bk1.v, s1);
      bk0.h[0] = *(const v8h*)(kr0 + 32); bk0.h[1] = *(const v8h*)(kr0 + 48);
      bk1.h[0] = *(const v8h*)(kr1 + 32); bk1.h[1] = *(const v8h*)(kr1 + 48);
      s0 = wmma_f16(aq1.v, bk0.v, s0);
      s1 = wmma_f16(aq1.v, bk1.v, s1);
    }

    // ---- online softmax (per-row stats, 16-lane reductions) ----
#pragma unroll
    for (int r = 0; r < 8; ++r) {
      float a0 = s0[r] * sc, a1 = s1[r] * sc;
      float mx = fmaxf(a0, a1);
#pragma unroll
      for (int off = 1; off < 16; off <<= 1)
        mx = fmaxf(mx, __shfl_xor(mx, off, 32));
      float mnew  = fmaxf(mrow[r], mx);
      float alpha = __expf(mrow[r] - mnew);
      float p0 = __expf(a0 - mnew);
      float p1 = __expf(a1 - mnew);
      float rs = p0 + p1;
#pragma unroll
      for (int off = 1; off < 16; off <<= 1)
        rs += __shfl_xor(rs, off, 32);
      lrow[r] = lrow[r] * alpha + rs;
      mrow[r] = mnew;
      O[0][r] *= alpha; O[1][r] *= alpha; O[2][r] *= alpha; O[3][r] *= alpha;
      // stage P (C-layout -> LDS, row-major [16][32])
      pbuf[(r + 8 * hv) * 32 + l16]      = (_Float16)p0;
      pbuf[(r + 8 * hv) * 32 + 16 + l16] = (_Float16)p1;
    }
    __asm volatile("s_wait_dscnt 0" ::: "memory");

    // reload P in A-matrix 16x32 layout
    AF pa;
    pa.h[0] = *(const v8h*)(pbuf + l16 * 32 + hv * 8);
    pa.h[1] = *(const v8h*)(pbuf + l16 * 32 + 16 + hv * 8);

    // ---- O += P * V (V^T layout gives contiguous B fragments) ----
#pragma unroll
    for (int nt = 0; nt < 4; ++nt) {
      AF bvf;
      const _Float16* vr = vb + (size_t)(nt * 16 + l16) * Sc + t0 + hv * 8;
      bvf.h[0] = *(const v8h*)(vr);
      bvf.h[1] = *(const v8h*)(vr + 16);
      O[nt] = wmma_f16(pa.v, bvf.v, O[nt]);
    }
  }

  // ---- epilogue: normalize and store heads [BH,S,E] f16 ----
  _Float16* hb = heads + (size_t)bh * Sc * Ec;
#pragma unroll
  for (int r = 0; r < 8; ++r) {
    float inv = 1.0f / lrow[r];
    int row = m0 + r + 8 * hv;
#pragma unroll
    for (int nt = 0; nt < 4; ++nt)
      hb[(size_t)row * Ec + nt * 16 + l16] = (_Float16)(O[nt][r] * inv);
  }
}

// ---------------- kernel 5: output projection + bias + residual ----------
// out[BS, D] = concat(heads) * Wo^T + bo + x     (f32 output)
__global__ __launch_bounds__(256)
void proj_out(const _Float16* __restrict__ heads, const _Float16* __restrict__ Wo,
              const float* __restrict__ bo, const float* __restrict__ x,
              float* __restrict__ out) {
  int lane = threadIdx.x & 31;
  int wave = threadIdx.x >> 5;
  int tile = blockIdx.x * 8 + wave;
  int m0 = (tile >> 4) * 16;
  int n0 = (tile & 15) * 64;
  int hv  = lane >> 4;
  int l16 = lane & 15;

  int arow = m0 + l16;
  int b = arow / Sc, s = arow % Sc;
  const _Float16* hbase = heads + (size_t)b * Hc * Sc * Ec + (size_t)s * Ec;

  v8f acc[4] = {v8f{}, v8f{}, v8f{}, v8f{}};
  for (int k0 = 0; k0 < Dc; k0 += 32) {
    int c0 = k0 + hv * 8;       // 8-half chunk, stays within one head (64)
    int c1 = k0 + 16 + hv * 8;
    AF a;
    a.h[0] = *(const v8h*)(hbase + (size_t)(c0 >> 6) * Sc * Ec + (c0 & 63));
    a.h[1] = *(const v8h*)(hbase + (size_t)(c1 >> 6) * Sc * Ec + (c1 & 63));
#pragma unroll
    for (int ns = 0; ns < 4; ++ns) {
      AF bm;
      const _Float16* wr = Wo + (size_t)(n0 + ns * 16 + l16) * Dc + k0 + hv * 8;
      bm.h[0] = *(const v8h*)(wr);
      bm.h[1] = *(const v8h*)(wr + 16);
      acc[ns] = wmma_f16(a.v, bm.v, acc[ns]);
    }
  }
#pragma unroll
  for (int ns = 0; ns < 4; ++ns) {
    int ncol = n0 + ns * 16 + l16;
    float bsv = bo[ncol];
#pragma unroll
    for (int r = 0; r < 8; ++r) {
      int row = m0 + r + 8 * hv;
      size_t idx = (size_t)row * Dc + ncol;
      out[idx] = acc[ns][r] + bsv + x[idx];
    }
  }
}

// ---------------- launcher ----------------
extern "C" void kernel_launch(void* const* d_in, const int* in_sizes, int n_in,
                              void* d_out, int out_size, void* d_ws, size_t ws_size,
                              hipStream_t stream) {
  const float* x     = (const float*)d_in[0];
  const float* Wq    = (const float*)d_in[1];
  const float* bq    = (const float*)d_in[2];
  const float* Wk    = (const float*)d_in[3];
  const float* bk    = (const float*)d_in[4];
  const float* Wv    = (const float*)d_in[5];
  const float* bv    = (const float*)d_in[6];
  const float* Wo    = (const float*)d_in[7];
  const float* bo    = (const float*)d_in[8];
  const float* gamma = (const float*)d_in[9];
  const float* beta  = (const float*)d_in[10];
  float* out = (float*)d_out;

  char* ws = (char*)d_ws;
  _Float16* xn_h    = (_Float16*)ws; ws += (size_t)BSc * Dc * 2;        // 16 MB
  _Float16* Wq_h    = (_Float16*)ws; ws += (size_t)Dc * Dc * 2;         // 2 MB
  _Float16* Wk_h    = (_Float16*)ws; ws += (size_t)Dc * Dc * 2;
  _Float16* Wv_h    = (_Float16*)ws; ws += (size_t)Dc * Dc * 2;
  _Float16* Wo_h    = (_Float16*)ws; ws += (size_t)Dc * Dc * 2;
  _Float16* q_h     = (_Float16*)ws; ws += (size_t)BHc * Sc * Ec * 2;   // 16 MB
  _Float16* k_h     = (_Float16*)ws; ws += (size_t)BHc * Sc * Ec * 2;
  _Float16* vT_h    = (_Float16*)ws; ws += (size_t)BHc * Ec * Sc * 2;
  _Float16* heads_h = (_Float16*)ws; ws += (size_t)BHc * Sc * Ec * 2;

  const int nW = Dc * Dc;
  cvt_f32_to_f16<<<(nW + 255) / 256, 256, 0, stream>>>(Wq, Wq_h, nW);
  cvt_f32_to_f16<<<(nW + 255) / 256, 256, 0, stream>>>(Wk, Wk_h, nW);
  cvt_f32_to_f16<<<(nW + 255) / 256, 256, 0, stream>>>(Wv, Wv_h, nW);
  cvt_f32_to_f16<<<(nW + 255) / 256, 256, 0, stream>>>(Wo, Wo_h, nW);

  ln_kernel<<<BSc, 256, 0, stream>>>(x, gamma, beta, xn_h);

  const int gemm_blocks = (BSc / 16) * (Dc / 64) / 8;   // 1024
  proj_qkv<<<gemm_blocks, 256, 0, stream>>>(xn_h, Wq_h, bq, q_h, 0);
  proj_qkv<<<gemm_blocks, 256, 0, stream>>>(xn_h, Wk_h, bk, k_h, 0);
  proj_qkv<<<gemm_blocks, 256, 0, stream>>>(xn_h, Wv_h, bv, vT_h, 1);

  flash_attn<<<BHc * (Sc / 16), 32, 0, stream>>>(q_h, k_h, vT_h, heads_h);

  proj_out<<<gemm_blocks, 256, 0, stream>>>(heads_h, Wo_h, bo, x, out);
}
